// GraphAttentionNetwork_39805756900100
// MI455X (gfx1250) — matrix operand order repaired and checked by reading, not compiled, not run on hardware
//
#include <hip/hip_runtime.h>
#include <cstdint>
#include <cstddef>

// ---------------------------------------------------------------------------
// Types for CDNA5 WMMA (wave32): v_wmma_f32_16x16x32_bf16
// ---------------------------------------------------------------------------
typedef __attribute__((ext_vector_type(16))) unsigned short v16u;
typedef __attribute__((ext_vector_type(8)))  unsigned short v8u;
typedef __attribute__((ext_vector_type(16))) __bf16         v16bf;
typedef __attribute__((ext_vector_type(8)))  float          v8f;

#define DEV static __device__ __forceinline__

DEV unsigned short f2bf(float f) {
  unsigned u = __float_as_uint(f);
  u += 0x7FFFu + ((u >> 16) & 1u);          // round-to-nearest-even
  return (unsigned short)(u >> 16);
}
// monotonic float <-> uint mapping for atomicMax on floats (incl. negatives)
DEV unsigned encf(float f) {
  unsigned u = __float_as_uint(f);
  unsigned mask = (u & 0x80000000u) ? 0xFFFFFFFFu : 0x80000000u;
  return u ^ mask;
}
DEV float decf(unsigned e) {
  unsigned mask = (e & 0x80000000u) ? 0x80000000u : 0xFFFFFFFFu;
  return __uint_as_float(e ^ mask);
}

// ---------------------------------------------------------------------------
// Utility kernels
// ---------------------------------------------------------------------------
__global__ void k_fill_u32(unsigned* __restrict__ p, unsigned v, long long n) {
  long long i = (long long)blockIdx.x * blockDim.x + threadIdx.x;
  if (i < n) p[i] = v;
}

__global__ void k_f32_to_bf16(const float* __restrict__ in,
                              unsigned short* __restrict__ out, long long n) {
  long long i = (long long)blockIdx.x * blockDim.x + threadIdx.x;
  if (i < n) out[i] = f2bf(in[i]);
}

// Pack row-major W[K x N] (f32) into per-lane bf16 B-fragments:
// Bp[((ks*Ntiles + nt)*32 + lane)*16 + i] = bf16(W[(ks*32 + (lane>=16?16:0) + i)*N + nt*16 + lane%16])
__global__ void k_pack_w(const float* __restrict__ W,
                         unsigned short* __restrict__ Bp, int K, int N) {
  int Ntiles = N >> 4;
  long long total = (long long)K * N;
  long long idx = (long long)blockIdx.x * blockDim.x + threadIdx.x;
  if (idx >= total) return;
  int i    = (int)(idx & 15);
  int lane = (int)((idx >> 4) & 31);
  long long t = idx >> 9;
  int nt = (int)(t % Ntiles);
  int ks = (int)(t / Ntiles);
  int k = ks * 32 + ((lane >> 4) << 4) + i;     // +16 for lanes 16..31
  int n = nt * 16 + (lane & 15);
  Bp[idx] = f2bf(W[(long long)k * N + n]);
}

// ---------------------------------------------------------------------------
// WMMA GEMM: C[MxN] (f32) = A[MxK] (bf16, row-major) * Bp (pre-packed bf16)
// One wave computes a 16 x (16*NT) strip. Per k-step: load the A fragment and
// ALL NT B fragments first (clause-able), then issue NT back-to-back WMMAs.
// ---------------------------------------------------------------------------
template <int NT>
__global__ __launch_bounds__(256)
void k_gemm_bf16(const unsigned short* __restrict__ A,
                 const unsigned short* __restrict__ Bp,
                 float* __restrict__ C, int M, int N, int K) {
  int lane = threadIdx.x & 31;
  int wave = threadIdx.x >> 5;
  int wavesPerBlk = blockDim.x >> 5;
  int Ntiles  = N >> 4;
  int Ngroups = Ntiles / NT;
  int Mtiles  = M >> 4;
  int tile = blockIdx.x * wavesPerBlk + wave;
  if (tile >= Mtiles * Ngroups) return;
  int mt  = tile / Ngroups;
  int ng  = tile - mt * Ngroups;
  int nt0 = ng * NT;
  int half = lane >> 4;     // 0: lanes 0-15, 1: lanes 16-31
  int lrow = lane & 15;

  // A fragment: lane holds row (mt*16+lrow); elems 0-15 = K off+0..15 (off = half*16)
  const unsigned short* Arow = A + (size_t)(mt * 16 + lrow) * K + half * 16;
  // B fragment: one contiguous 16-element (32B) chunk per lane per k-step per tile
  const unsigned short* Bbase = Bp + ((size_t)nt0 * 32 + lane) * 16;
  size_t bstride = (size_t)Ntiles * 512;   // elements per k-step

  v8f acc[NT];
#pragma unroll
  for (int t = 0; t < NT; ++t) acc[t] = (v8f){};

  int Ksteps = K >> 5;
  for (int ks = 0; ks < Ksteps; ++ks) {
    // --- issue all loads for this k-step up front ---
    v8u alo = *(const v8u*)(Arow + ks * 32);
    v8u ahi = *(const v8u*)(Arow + ks * 32 + 8);
    v16u bv[NT];
#pragma unroll
    for (int t = 0; t < NT; ++t)
      bv[t] = *(const v16u*)(Bbase + (size_t)ks * bstride + (size_t)t * 512);

    v16u av;
#pragma unroll
    for (int j = 0; j < 8; ++j) { av[j] = alo[j]; av[j + 8] = ahi[j]; }
    v16bf a = __builtin_bit_cast(v16bf, av);

    // --- NT independent WMMAs, back-to-back ---
#pragma unroll
    for (int t = 0; t < NT; ++t)
      acc[t] = __builtin_amdgcn_wmma_f32_16x16x32_bf16(
          false, a, false, __builtin_bit_cast(v16bf, bv[t]),
          (short)0, acc[t], false, false);
  }
  // C layout: acc[t][r] -> row mt*16 + half*8 + r, col (nt0+t)*16 + lrow
  int row0 = mt * 16 + half * 8;
#pragma unroll
  for (int t = 0; t < NT; ++t) {
    int col = (nt0 + t) * 16 + lrow;
#pragma unroll
    for (int r = 0; r < 8; ++r) C[(size_t)(row0 + r) * N + col] = acc[t][r];
  }
}

// ---------------------------------------------------------------------------
// Attention coefficient sums: a_src[n,h] = sum_c h[n,h,c]*att_src[h,c]
// ---------------------------------------------------------------------------
__global__ void k_att(const float* __restrict__ h, const float* __restrict__ att_s,
                      const float* __restrict__ att_d,
                      float* __restrict__ as, float* __restrict__ ad, int N, int H) {
  int gid = blockIdx.x * blockDim.x + threadIdx.x;
  if (gid >= N * H) return;
  int n = gid / H, hh = gid - n * H;
  const float* hp = h + ((size_t)n * H + hh) * 32;
  const float* sp = att_s + hh * 32;
  const float* dp = att_d + hh * 32;
  float ss = 0.f, dd = 0.f;
#pragma unroll
  for (int c = 0; c < 32; ++c) { float v = hp[c]; ss += v * sp[c]; dd += v * dp[c]; }
  as[gid] = ss; ad[gid] = dd;
}

// ---------------------------------------------------------------------------
// Edge passes (Et = E real edges + N self loops)
// ---------------------------------------------------------------------------
DEV void edge_sd(const int* ei, int E, int eid, int& s, int& d) {
  if (eid < E) { s = ei[eid]; d = ei[E + eid]; }
  else         { s = eid - E; d = eid - E; }
}

template <int H>
__global__ void k_edge_max(const int* __restrict__ ei, int E, int N,
                           const float* __restrict__ as, const float* __restrict__ ad,
                           unsigned* __restrict__ mx) {
  long long gid = (long long)blockIdx.x * blockDim.x + threadIdx.x;
  if (gid >= (long long)(E + N) * H) return;
  int eid = (int)(gid / H), hh = (int)(gid - (long long)eid * H);
  int s, d; edge_sd(ei, E, eid, s, d);
  float e = as[(size_t)s * H + hh] + ad[(size_t)d * H + hh];
  e = e > 0.f ? e : 0.2f * e;                      // leaky_relu(0.2)
  atomicMax(&mx[(size_t)d * H + hh], encf(e));
}

template <int H>
__global__ void k_edge_den(const int* __restrict__ ei, int E, int N,
                           const float* __restrict__ as, const float* __restrict__ ad,
                           const unsigned* __restrict__ mx, float* __restrict__ den) {
  long long gid = (long long)blockIdx.x * blockDim.x + threadIdx.x;
  if (gid >= (long long)(E + N) * H) return;
  int eid = (int)(gid / H), hh = (int)(gid - (long long)eid * H);
  int s, d; edge_sd(ei, E, eid, s, d);
  size_t di = (size_t)d * H + hh;
  float e = as[(size_t)s * H + hh] + ad[di];
  e = e > 0.f ? e : 0.2f * e;
  atomicAdd(&den[di], expf(e - decf(mx[di])));
}

template <int H>
__global__ void k_edge_msg(const int* __restrict__ ei, int E, int N,
                           const float* __restrict__ as, const float* __restrict__ ad,
                           const unsigned* __restrict__ mx, const float* __restrict__ den,
                           const float* __restrict__ hfeat, float* __restrict__ out) {
  long long gid = (long long)blockIdx.x * blockDim.x + threadIdx.x;
  if (gid >= (long long)(E + N) * H) return;
  int eid = (int)(gid / H), hh = (int)(gid - (long long)eid * H);
  int s, d; edge_sd(ei, E, eid, s, d);
  size_t di = (size_t)d * H + hh;
  float e = as[(size_t)s * H + hh] + ad[di];
  e = e > 0.f ? e : 0.2f * e;
  float alpha = expf(e - decf(mx[di])) / (den[di] + 1e-16f);
  const float* hp = hfeat + ((size_t)s * H + hh) * 32;
  float* op = out + di * 32;
#pragma unroll
  for (int c = 0; c < 32; ++c) atomicAdd(&op[c], hp[c] * alpha);
}

// ---------------------------------------------------------------------------
// Bias + ELU (+ bf16 conversion for next GEMM input)
// ---------------------------------------------------------------------------
__global__ void k_bias_elu_bf16(const float* __restrict__ in, const float* __restrict__ b,
                                unsigned short* __restrict__ out, long long n, int F) {
  long long i = (long long)blockIdx.x * blockDim.x + threadIdx.x;
  if (i >= n) return;
  float v = in[i] + b[(int)(i % F)];
  v = v > 0.f ? v : expm1f(v);
  out[i] = f2bf(v);
}

__global__ void k_bias_elu_f32(const float* __restrict__ in, const float* __restrict__ b,
                               float* __restrict__ out, long long n, int F) {
  long long i = (long long)blockIdx.x * blockDim.x + threadIdx.x;
  if (i >= n) return;
  float v = in[i] + b[(int)(i % F)];
  out[i] = v > 0.f ? v : expm1f(v);
}

// ---------------------------------------------------------------------------
// Global mean pool + final dense + sigmoid
// ---------------------------------------------------------------------------
__global__ void k_pool(const float* __restrict__ h, const int* __restrict__ batch,
                       float* __restrict__ psum, float* __restrict__ pcnt, int N) {
  int gid = blockIdx.x * blockDim.x + threadIdx.x;
  if (gid >= N * 32) return;
  int n = gid >> 5, c = gid & 31;
  int g = batch[n];
  atomicAdd(&psum[(size_t)g * 32 + c], h[gid]);
  if (c == 0) atomicAdd(&pcnt[g], 1.0f);
}

__global__ void k_final(const float* __restrict__ psum, const float* __restrict__ pcnt,
                        const float* __restrict__ Wo, const float* __restrict__ bo,
                        float* __restrict__ out, int G) {
  int g = blockIdx.x * blockDim.x + threadIdx.x;
  if (g >= G) return;
  float inv = 1.0f / fmaxf(pcnt[g], 1.0f);
  float s = bo[0];
#pragma unroll
  for (int c = 0; c < 32; ++c) s += psum[(size_t)g * 32 + c] * inv * Wo[c];
  out[g] = 1.0f / (1.0f + expf(-s));
}

// ---------------------------------------------------------------------------
// Host-side launch
// ---------------------------------------------------------------------------
static inline unsigned gridFor(long long n, int b) { return (unsigned)((n + b - 1) / b); }

extern "C" void kernel_launch(void* const* d_in, const int* in_sizes, int n_in,
                              void* d_out, int out_size, void* d_ws, size_t ws_size,
                              hipStream_t stream) {
  const float* x        = (const float*)d_in[0];
  const int*   ei       = (const int*)d_in[1];
  const int*   batch    = (const int*)d_in[2];
  const float* W1       = (const float*)d_in[3];
  const float* att_src1 = (const float*)d_in[4];
  const float* att_dst1 = (const float*)d_in[5];
  const float* b1       = (const float*)d_in[6];
  const float* W2       = (const float*)d_in[7];
  const float* att_src2 = (const float*)d_in[8];
  const float* att_dst2 = (const float*)d_in[9];
  const float* b2       = (const float*)d_in[10];
  const float* Wo       = (const float*)d_in[11];
  const float* bo       = (const float*)d_in[12];

  const int FIN = 128, HID = 32, HEADS = 8, G = 64;
  const int N = in_sizes[0] / FIN;          // 50000
  const int E = in_sizes[1] / 2;            // 800000
  const int F1 = HEADS * HID;               // 256
  const long long Et = (long long)E + N;

  // ---- workspace layout (bump allocator with region reuse) ----
  char* ws = (char*)d_ws;
  size_t off = 0;
  auto take = [&](size_t bytes) { size_t r = off; off += (bytes + 255) & ~(size_t)255; return r; };

  size_t SZ_XB  = (size_t)N * FIN * 2;        // 12.8 MB (bf16 x)  | reused by h2 + out2
  size_t SZ_H1  = (size_t)N * F1 * 4;         // 51.2 MB (h1 f32)  | reused by h1b + L2 small bufs
  size_t oA = take(SZ_XB);
  size_t oB = take(SZ_H1);
  size_t oASRC1 = take((size_t)N * HEADS * 4);
  size_t oADST1 = take((size_t)N * HEADS * 4);
  size_t oMAX1  = take((size_t)N * HEADS * 4);
  size_t oDEN1  = take((size_t)N * HEADS * 4);
  size_t oOUT1  = take((size_t)N * F1 * 4);
  size_t oW1P   = take((size_t)FIN * F1 * 2);
  size_t oW2P   = take((size_t)F1 * HID * 2);

  unsigned short* xb  = (unsigned short*)(ws + oA);
  float* h1           = (float*)(ws + oB);
  float* asrc1        = (float*)(ws + oASRC1);
  float* adst1        = (float*)(ws + oADST1);
  unsigned* max1      = (unsigned*)(ws + oMAX1);
  float* den1         = (float*)(ws + oDEN1);
  float* out1         = (float*)(ws + oOUT1);
  unsigned short* w1p = (unsigned short*)(ws + oW1P);
  unsigned short* w2p = (unsigned short*)(ws + oW2P);

  // region B reuse (h1 dead after L1 message pass):
  unsigned short* h1b = (unsigned short*)(ws + oB);                     // 25.6 MB
  size_t oB2 = oB + (((size_t)N * F1 * 2 + 255) & ~(size_t)255);
  float* asrc2  = (float*)(ws + oB2);                 oB2 += ((size_t)N * 4 + 255) & ~(size_t)255;
  float* adst2  = (float*)(ws + oB2);                 oB2 += ((size_t)N * 4 + 255) & ~(size_t)255;
  unsigned* max2= (unsigned*)(ws + oB2);              oB2 += ((size_t)N * 4 + 255) & ~(size_t)255;
  float* den2   = (float*)(ws + oB2);                 oB2 += ((size_t)N * 4 + 255) & ~(size_t)255;
  float* h2f    = (float*)(ws + oB2);                 oB2 += ((size_t)N * HID * 4 + 255) & ~(size_t)255;
  float* psum   = (float*)(ws + oB2);                 oB2 += ((size_t)G * HID * 4 + 255) & ~(size_t)255;
  float* pcnt   = (float*)(ws + oB2);
  // region A reuse (xb dead after GEMM1):
  float* h2   = (float*)(ws + oA);
  float* out2 = (float*)(ws + oA + (((size_t)N * HID * 4 + 255) & ~(size_t)255));

  const int B = 256;

  // ===== Layer 1 =====
  k_f32_to_bf16<<<gridFor((long long)N * FIN, B), B, 0, stream>>>(x, xb, (long long)N * FIN);
  k_pack_w<<<gridFor((long long)FIN * F1, B), B, 0, stream>>>(W1, w1p, FIN, F1);
  k_pack_w<<<gridFor((long long)F1 * HID, B), B, 0, stream>>>(W2, w2p, F1, HID);

  {
    long long waves = (long long)(N >> 4) * ((F1 >> 4) / 4);   // NT=4
    k_gemm_bf16<4><<<gridFor(waves, 8), B, 0, stream>>>(xb, w1p, h1, N, F1, FIN);
  }
  k_att<<<gridFor((long long)N * HEADS, B), B, 0, stream>>>(h1, att_src1, att_dst1, asrc1, adst1, N, HEADS);

  k_fill_u32<<<gridFor((long long)N * HEADS, B), B, 0, stream>>>(max1, 0u, (long long)N * HEADS);
  k_fill_u32<<<gridFor((long long)N * HEADS, B), B, 0, stream>>>((unsigned*)den1, 0u, (long long)N * HEADS);
  k_fill_u32<<<gridFor((long long)N * F1, B), B, 0, stream>>>((unsigned*)out1, 0u, (long long)N * F1);

  k_edge_max<8><<<gridFor(Et * HEADS, B), B, 0, stream>>>(ei, E, N, asrc1, adst1, max1);
  k_edge_den<8><<<gridFor(Et * HEADS, B), B, 0, stream>>>(ei, E, N, asrc1, adst1, max1, den1);
  k_edge_msg<8><<<gridFor(Et * HEADS, B), B, 0, stream>>>(ei, E, N, asrc1, adst1, max1, den1, h1, out1);

  k_bias_elu_bf16<<<gridFor((long long)N * F1, B), B, 0, stream>>>(out1, b1, h1b, (long long)N * F1, F1);

  // ===== Layer 2 =====
  {
    long long waves = (long long)(N >> 4) * ((HID >> 4) / 2);  // NT=2
    k_gemm_bf16<2><<<gridFor(waves, 8), B, 0, stream>>>(h1b, w2p, h2, N, HID, F1);
  }
  k_att<<<gridFor((long long)N, B), B, 0, stream>>>(h2, att_src2, att_dst2, asrc2, adst2, N, 1);

  k_fill_u32<<<gridFor((long long)N, B), B, 0, stream>>>(max2, 0u, (long long)N);
  k_fill_u32<<<gridFor((long long)N, B), B, 0, stream>>>((unsigned*)den2, 0u, (long long)N);
  k_fill_u32<<<gridFor((long long)N * HID, B), B, 0, stream>>>((unsigned*)out2, 0u, (long long)N * HID);
  k_fill_u32<<<gridFor((long long)G * HID, B), B, 0, stream>>>((unsigned*)psum, 0u, (long long)G * HID);
  k_fill_u32<<<gridFor((long long)G, B), B, 0, stream>>>((unsigned*)pcnt, 0u, (long long)G);

  k_edge_max<1><<<gridFor(Et, B), B, 0, stream>>>(ei, E, N, asrc2, adst2, max2);
  k_edge_den<1><<<gridFor(Et, B), B, 0, stream>>>(ei, E, N, asrc2, adst2, max2, den2);
  k_edge_msg<1><<<gridFor(Et, B), B, 0, stream>>>(ei, E, N, asrc2, adst2, max2, den2, h2, out2);

  k_bias_elu_f32<<<gridFor((long long)N * HID, B), B, 0, stream>>>(out2, b2, h2f, (long long)N * HID, HID);

  // ===== Pool + readout =====
  k_pool<<<gridFor((long long)N * HID, B), B, 0, stream>>>(h2f, batch, psum, pcnt, N);
  k_final<<<1, 64, 0, stream>>>(psum, pcnt, Wo, bo, (float*)d_out, G);
}
